// GCN_90529320665180
// MI455X (gfx1250) — matrix operand order repaired and checked by reading, not compiled, not run on hardware
//
#include <hip/hip_runtime.h>
#include <stdint.h>

// Problem sizes (fixed by the reference)
#define BB 64
#define TT 512
#define DD 1024
#define MM (BB * TT)
#define SCALE 0.03125f          // 1/sqrt(1024)
#define POOL_NEG (-1e12f)

typedef __attribute__((ext_vector_type(16))) __bf16 v16bf;
typedef __attribute__((ext_vector_type(8)))  float  v8f;
typedef __attribute__((ext_vector_type(4)))  unsigned int u32x4;
typedef __attribute__((ext_vector_type(8)))  int    i32x8;
typedef __attribute__((ext_vector_type(4)))  int    i32x4;

struct Pack32 { uint4 lo, hi; };

__device__ __forceinline__ v16bf make_frag(uint4 lo, uint4 hi) {
    Pack32 p; p.lo = lo; p.hi = hi;
    return __builtin_bit_cast(v16bf, p);
}

__device__ __forceinline__ __bf16 f2bf(float f) {
    uint32_t u = __builtin_bit_cast(uint32_t, f);
    u += 0x7FFFu + ((u >> 16) & 1u);          // round-to-nearest-even
    uint16_t h = (uint16_t)(u >> 16);
    return __builtin_bit_cast(__bf16, h);
}

__device__ __forceinline__ v8f wmma_bf16f32(v16bf a, v16bf b, v8f c) {
    // (neg_a, A, neg_b, B, c_mod, C, reuse_a, reuse_b)
    return __builtin_amdgcn_wmma_f32_16x16x32_bf16(false, a, false, b,
                                                   (short)0, c, false, false);
}

// ---------------------------------------------------------------------------
// TDM: DMA a 2D tile (128 rows x 32 bf16 elems, row stride 1024 elems) from
// global memory into LDS. D# layout per CDNA5 ISA §8 (CDNA5 ISA).
// Issued by one wave; completion tracked with TENSORcnt.
// This toolchain's builtin is the 6-arg form:
//   (u32x4 g0, i32x8 g1, i32x4 g2, i32x4 g3, i32x8 extra, i32 cpol)
// ---------------------------------------------------------------------------
__device__ __forceinline__ void tdm_load_tile_2d(
    uint32_t lds_off, const void* gsrc, uint32_t nrows_total) {
    const uint64_t ga = (uint64_t)(uintptr_t)gsrc;
    // group0: [1:0]=count=1 | [63:32]=lds_addr | [120:64]=global_addr |
    //         [127:126]=type=2
    u32x4 g0 = { 1u,
                 lds_off,
                 (uint32_t)ga,
                 (uint32_t)(ga >> 32) | (2u << 30) };
    // group1: data_size=1(2B); tensor_dim0=1024; tensor_dim1=nrows_total;
    //         tile_dim0=32; tile_dim1=128; tensor_dim0_stride=1024
    i32x8 g1 = { (int)(1u << 16),                                  // dword0
                 (int)((1024u & 0xffffu) << 16),                   // dword1
                 (int)((1024u >> 16) |
                       ((nrows_total & 0xffffu) << 16)),           // dword2
                 (int)((nrows_total >> 16) | (32u << 16)),         // dword3
                 128,                                              // dword4
                 1024,                                             // dword5
                 0, 0 };                                           // dword6/7
    i32x4 z4 = { 0, 0, 0, 0 };                 // 2D tensor: groups 2/3 unused
    i32x8 z8 = { 0, 0, 0, 0, 0, 0, 0, 0 };
    __builtin_amdgcn_tensor_load_to_lds(g0, g1, z4, z4, z8, 0);
}

// ---------------------------------------------------------------------------
// 0. fused pass over X: bf16 convert (for WMMA GEMMs) + the two pass-through
//    outputs (gcn_inputs copy, masks int->float)
// ---------------------------------------------------------------------------
__global__ __launch_bounds__(256) void xcvt_copy_kernel(
    const float* __restrict__ x, const int* __restrict__ masks,
    __bf16* __restrict__ xb, float* __restrict__ out_mask,
    float* __restrict__ out_x) {
    const size_t tid  = (size_t)blockIdx.x * 256 + threadIdx.x;
    const size_t base = tid * 8;
    const float4* s = (const float4*)(x + base);
    float4 f0 = s[0], f1 = s[1];
    union { __bf16 h[8]; uint4 q; } t;
    t.h[0] = f2bf(f0.x); t.h[1] = f2bf(f0.y);
    t.h[2] = f2bf(f0.z); t.h[3] = f2bf(f0.w);
    t.h[4] = f2bf(f1.x); t.h[5] = f2bf(f1.y);
    t.h[6] = f2bf(f1.z); t.h[7] = f2bf(f1.w);
    *(uint4*)(xb + base) = t.q;
    ((float4*)(out_x + base))[0] = f0;
    ((float4*)(out_x + base))[1] = f1;
    if (tid < (size_t)BB * TT) out_mask[tid] = (float)masks[tid];
}

// ---------------------------------------------------------------------------
// 1. max-pool over subject span -> subj_obj [B, 2D]
// ---------------------------------------------------------------------------
__global__ __launch_bounds__(256) void pool_kernel(
    const float* __restrict__ x, const int* __restrict__ subj_pos,
    float* __restrict__ subj_obj) {
    const int b = blockIdx.x;
    __shared__ int sp[TT];
    for (int t = threadIdx.x; t < TT; t += 256) sp[t] = subj_pos[b * TT + t];
    __syncthreads();
    for (int d = threadIdx.x; d < DD; d += 256) {
        float mv = POOL_NEG;
        const float* xb = x + (size_t)b * TT * DD + d;
        for (int t = 0; t < TT; ++t)
            if (sp[t] == 0) mv = fmaxf(mv, xb[(size_t)t * DD]);
        subj_obj[b * 2 * DD + d]      = mv;
        subj_obj[b * 2 * DD + DD + d] = mv;
    }
}

// ---------------------------------------------------------------------------
// 2. q = relu(subj_obj @ Wq + bq)
// ---------------------------------------------------------------------------
__global__ __launch_bounds__(256) void q_kernel(
    const float* __restrict__ so, const float* __restrict__ Wq,
    const float* __restrict__ bq, float* __restrict__ q) {
    const int b = blockIdx.y;
    const int n = blockIdx.x * 256 + threadIdx.x;
    float acc = bq[n];
    const float* sb = so + b * 2 * DD;
    for (int k = 0; k < 2 * DD; ++k) acc += sb[k] * Wq[(size_t)k * DD + n];
    q[b * DD + n] = fmaxf(acc, 0.f);
}

// ---------------------------------------------------------------------------
// 3. tvec = relu([q,0] @ Wc + bc); tw = tvec * Wk
// ---------------------------------------------------------------------------
__global__ __launch_bounds__(256) void tw_kernel(
    const float* __restrict__ q, const float* __restrict__ Wc,
    const float* __restrict__ bc, const float* __restrict__ Wk,
    float* __restrict__ tw) {
    const int b = blockIdx.y;
    const int n = blockIdx.x * 256 + threadIdx.x;
    float acc = bc[n];
    const float* qb = q + b * DD;
    for (int k = 0; k < DD; ++k) acc += qb[k] * Wc[(size_t)k * DD + n];
    tw[b * DD + n] = fmaxf(acc, 0.f) * Wk[n];
}

// ---------------------------------------------------------------------------
// 4. klogit[b,t] = sum_d tw[b,d] * x[b,t,d] + bk   (one wave per (b,t))
// ---------------------------------------------------------------------------
__global__ __launch_bounds__(256) void klogit_kernel(
    const float* __restrict__ x, const float* __restrict__ tw,
    const float* __restrict__ bk, float* __restrict__ klog) {
    const int wave = threadIdx.x >> 5, lane = threadIdx.x & 31;
    const int g = blockIdx.x * 8 + wave;        // flat (b*T + t)
    const int b = g >> 9;
    const float* xr  = x  + (size_t)g * DD;
    const float* twr = tw + (size_t)b * DD;
    float p = 0.f;
    for (int d = lane; d < DD; d += 32) p += twr[d] * xr[d];
    for (int off = 16; off > 0; off >>= 1) p += __shfl_xor(p, off, 32);
    if (lane == 0) klog[g] = p + bk[0];
}

// ---------------------------------------------------------------------------
// 5. kw = softmax_t(klogit)
// ---------------------------------------------------------------------------
__global__ __launch_bounds__(256) void softmaxT_kernel(
    const float* __restrict__ klog, float* __restrict__ kw) {
    const int b = blockIdx.x, tid = threadIdx.x;
    __shared__ float red[256];
    float mx = -3.4e38f;
    for (int t = tid; t < TT; t += 256) mx = fmaxf(mx, klog[b * TT + t]);
    red[tid] = mx; __syncthreads();
    for (int s = 128; s > 0; s >>= 1) {
        if (tid < s) red[tid] = fmaxf(red[tid], red[tid + s]);
        __syncthreads();
    }
    mx = red[0]; __syncthreads();
    float sm = 0.f;
    for (int t = tid; t < TT; t += 256) sm += __expf(klog[b * TT + t] - mx);
    red[tid] = sm; __syncthreads();
    for (int s = 128; s > 0; s >>= 1) {
        if (tid < s) red[tid] += red[tid + s];
        __syncthreads();
    }
    const float inv = 1.f / red[0];
    for (int t = tid; t < TT; t += 256)
        kw[b * TT + t] = __expf(klog[b * TT + t] - mx) * inv;
}

// ---------------------------------------------------------------------------
// 6. c[b,d] = sum_t kw[b,t] * x[b,t,d]
// ---------------------------------------------------------------------------
__global__ __launch_bounds__(256) void c_kernel(
    const float* __restrict__ x, const float* __restrict__ kw,
    float* __restrict__ c) {
    const int b = blockIdx.y;
    const int d = blockIdx.x * 256 + threadIdx.x;
    float acc = 0.f;
    const float* xb = x + (size_t)b * TT * DD + d;
    const float* kb = kw + b * TT;
    for (int t = 0; t < TT; ++t) acc += kb[t] * xb[(size_t)t * DD];
    c[b * DD + d] = acc;
}

// ---------------------------------------------------------------------------
// 7. m = relu([c, subj_obj] @ Wm + bm)
// ---------------------------------------------------------------------------
__global__ __launch_bounds__(256) void m_kernel(
    const float* __restrict__ c, const float* __restrict__ so,
    const float* __restrict__ Wm, const float* __restrict__ bm,
    float* __restrict__ m) {
    const int b = blockIdx.y;
    const int n = blockIdx.x * 256 + threadIdx.x;
    float acc = bm[n];
    for (int k = 0; k < DD; ++k)
        acc += c[b * DD + k] * Wm[(size_t)k * DD + n];
    for (int k = 0; k < 2 * DD; ++k)
        acc += so[b * 2 * DD + k] * Wm[(size_t)(DD + k) * DD + n];
    m[b * DD + n] = fmaxf(acc, 0.f);
}

// ---------------------------------------------------------------------------
// 8. weight transpose + bf16 convert: Wt[n][k] = bf16(W[k][n])  (DxD)
// ---------------------------------------------------------------------------
__global__ __launch_bounds__(256) void wt_kernel(
    const float* __restrict__ W, __bf16* __restrict__ Wt) {
    const size_t i = (size_t)blockIdx.x * 256 + threadIdx.x;   // over D*D
    const int n = (int)(i >> 10), k = (int)(i & (DD - 1));
    Wt[i] = f2bf(W[(size_t)k * DD + n]);
}

// ---------------------------------------------------------------------------
// 9. Tiled bf16 WMMA GEMM with TDM-fed, double-buffered LDS tiles:
//    C(bf16[M,N]) = A(bf16[M,1024]) x Wt(bf16[N,1024])^T + bias
//    Block tile 128x128x32; 8 waves (4x2); wave 0 drives the Tensor Data
//    Mover; compute waves only issue ds_load_b128 + v_wmma.
// ---------------------------------------------------------------------------
__global__ __launch_bounds__(256) void gemm_proj_kernel(
    const __bf16* __restrict__ A, const __bf16* __restrict__ Wt,
    const float* __restrict__ bias, __bf16* __restrict__ C) {
    __shared__ __bf16 As[2][128][32];   // 16 KB (double-buffered A tile)
    __shared__ __bf16 Bs[2][128][32];   // 16 KB (double-buffered W tile)
    __shared__ __bf16 Cs[128][128];     // 32 KB (epilogue restage)
    const int tid  = threadIdx.x;
    const int wave = tid >> 5, lane = tid & 31;
    const int lhalf = lane & 15, lsel = lane >> 4;
    const int wm = wave >> 1, wn = wave & 1;     // 4 x 2 wave grid
    const int m0 = blockIdx.y * 128, n0 = blockIdx.x * 128;

    const __bf16* Abase = A  + (size_t)m0 * DD;
    const __bf16* Bbase = Wt + (size_t)n0 * DD;
    const uint32_t offA0 = (uint32_t)(uintptr_t)&As[0][0][0];
    const uint32_t offA1 = (uint32_t)(uintptr_t)&As[1][0][0];
    const uint32_t offB0 = (uint32_t)(uintptr_t)&Bs[0][0][0];
    const uint32_t offB1 = (uint32_t)(uintptr_t)&Bs[1][0][0];

    v8f acc[2][4];
#pragma unroll
    for (int i = 0; i < 2; ++i)
#pragma unroll
        for (int j = 0; j < 4; ++j)
#pragma unroll
            for (int e = 0; e < 8; ++e) acc[i][j][e] = 0.f;

    // prologue: DMA first tiles into buffer 0
    if (wave == 0) {
        tdm_load_tile_2d(offA0, Abase, (uint32_t)MM);
        tdm_load_tile_2d(offB0, Bbase, (uint32_t)DD);
        __builtin_amdgcn_s_wait_tensorcnt(0);
    }
    __syncthreads();

    for (int k0 = 0; k0 < DD; k0 += 32) {
        const int buf = (k0 >> 5) & 1;
        const bool more = (k0 + 32 < DD);
        if (wave == 0 && more) {       // async DMA of next tiles (other buf)
            tdm_load_tile_2d(buf ? offA0 : offA1, Abase + k0 + 32,
                             (uint32_t)MM);
            tdm_load_tile_2d(buf ? offB0 : offB1, Bbase + k0 + 32,
                             (uint32_t)DD);
        }

        v16bf af[2], bfr[4];
#pragma unroll
        for (int mt = 0; mt < 2; ++mt) {
            const __bf16* ap = &As[buf][wm * 32 + mt * 16 + lhalf][lsel * 8];
            af[mt] = make_frag(*(const uint4*)ap, *(const uint4*)(ap + 16));
        }
#pragma unroll
        for (int nt = 0; nt < 4; ++nt) {
            const __bf16* bp = &Bs[buf][wn * 64 + nt * 16 + lhalf][lsel * 8];
            bfr[nt] = make_frag(*(const uint4*)bp, *(const uint4*)(bp + 16));
        }
#pragma unroll
        for (int mt = 0; mt < 2; ++mt)
#pragma unroll
            for (int nt = 0; nt < 4; ++nt)
                acc[mt][nt] = wmma_bf16f32(af[mt], bfr[nt], acc[mt][nt]);

        if (wave == 0 && more) __builtin_amdgcn_s_wait_tensorcnt(0);
        __syncthreads();
    }

    // epilogue: bias add, bf16 restage in LDS, coalesced global store
#pragma unroll
    for (int mt = 0; mt < 2; ++mt)
#pragma unroll
        for (int nt = 0; nt < 4; ++nt) {
            const int col = wn * 64 + nt * 16 + lhalf;
            const float bv = bias[n0 + col];
#pragma unroll
            for (int v = 0; v < 8; ++v) {
                const int row = wm * 32 + mt * 16 + v + lsel * 8;
                Cs[row][col] = f2bf(acc[mt][nt][v] + bv);
            }
        }
    __syncthreads();
    for (int i = tid; i < 128 * 128 / 8; i += 256) {
        const int rr = i >> 4;
        const int cc = (i & 15) * 8;
        *(uint4*)(C + (size_t)(m0 + rr) * DD + n0 + cc) =
            *(const uint4*)&Cs[rr][cc];
    }
}

// ---------------------------------------------------------------------------
// 10. V transpose: Vt[b][d][t] = V[b][t][d]   (bf16)
// ---------------------------------------------------------------------------
__global__ __launch_bounds__(256) void vt_kernel(
    const __bf16* __restrict__ V, __bf16* __restrict__ Vt) {
    const size_t i = (size_t)blockIdx.x * 256 + threadIdx.x;   // b,d,t linear
    const int t = (int)(i & (TT - 1));
    const int d = (int)((i >> 9) & (DD - 1));
    const size_t b = i >> 19;
    Vt[i] = V[(b * TT + t) * DD + d];
}

// ---------------------------------------------------------------------------
// 11. Fused attention: scores -> dual softmax -> att + gated P*V output
// ---------------------------------------------------------------------------
__global__ __launch_bounds__(256) void attn_kernel(
    const __bf16* __restrict__ Kb,   // [B,T,D]
    const __bf16* __restrict__ Qb,   // [B,T,D]
    const __bf16* __restrict__ Vt,   // [B,D,T]
    const float*  __restrict__ gate, // m: [B,D]
    float* __restrict__ out,         // [B,T,D]
    float* __restrict__ att) {       // [B,T]
    __shared__ float  sc[16][TT];    // 32 KB raw scores
    __shared__ __bf16 pr[16][TT];    // 16 KB probs (scaled softmax), bf16
    __shared__ float  rmax[16];
    __shared__ float  rsum2[16];

    const int tid  = threadIdx.x;
    const int wave = tid >> 5, lane = tid & 31;
    const int lhalf = lane & 15, lsel = lane >> 4;
    const int t0 = blockIdx.x * 16;
    const int b  = blockIdx.y;

    // ---- phase 1: scores[16 x 512] = K_rows @ Q^T (WMMA, bf16) ----
    const __bf16* Kbase = Kb + ((size_t)b * TT + t0) * DD;
    const __bf16* Qbase = Qb + (size_t)b * TT * DD;
    const int s0 = wave * 64;                    // this wave's 64 score cols
    v8f acc[4];
#pragma unroll
    for (int j = 0; j < 4; ++j)
#pragma unroll
        for (int e = 0; e < 8; ++e) acc[j][e] = 0.f;

    for (int d0 = 0; d0 < DD; d0 += 32) {
        const __bf16* ap = Kbase + (size_t)lhalf * DD + d0 + lsel * 8;
        v16bf af = make_frag(*(const uint4*)ap, *(const uint4*)(ap + 16));
#pragma unroll
        for (int nt = 0; nt < 4; ++nt) {
            const __bf16* bp =
                Qbase + (size_t)(s0 + nt * 16 + lhalf) * DD + d0 + lsel * 8;
            v16bf bfr = make_frag(*(const uint4*)bp, *(const uint4*)(bp + 16));
            acc[nt] = wmma_bf16f32(af, bfr, acc[nt]);
        }
    }
#pragma unroll
    for (int nt = 0; nt < 4; ++nt)
#pragma unroll
        for (int v = 0; v < 8; ++v)
            sc[v + lsel * 8][s0 + nt * 16 + lhalf] = acc[nt][v];
    __syncthreads();

    // ---- phase 2: per-row stats; att[b,t] = scale * (1 - p_diag_raw) ----
    if (tid < 16) {
        float mx = -3.4e38f;
        for (int s = 0; s < TT; ++s) mx = fmaxf(mx, sc[tid][s]);
        float s1 = 0.f, s2 = 0.f;
        for (int s = 0; s < TT; ++s) {
            const float xv = sc[tid][s] - mx;
            s1 += __expf(xv);
            s2 += __expf(xv * SCALE);
        }
        const int trow = t0 + tid;
        const float ediag = __expf(sc[tid][trow] - mx);
        att[(size_t)b * TT + trow] = SCALE * (1.0f - ediag / s1);
        rmax[tid]  = mx;
        rsum2[tid] = s2;
    }
    __syncthreads();

    // ---- phase 3: probs (scaled softmax) -> bf16 in LDS ----
    for (int i = tid; i < 16 * TT; i += 256) {
        const int rr = i >> 9, ss = i & (TT - 1);
        const float p = __expf((sc[rr][ss] - rmax[rr]) * SCALE) / rsum2[rr];
        pr[rr][ss] = f2bf(p);
    }
    __syncthreads();

    // ---- phase 4: out[16 x 1024] = probs @ V, gated by m ----
    const __bf16* Vbase = Vt + (size_t)b * DD * TT;
    const int dbase = wave * 128;                 // this wave's 128 d-cols
    v8f oacc[8];
#pragma unroll
    for (int j = 0; j < 8; ++j)
#pragma unroll
        for (int e = 0; e < 8; ++e) oacc[j][e] = 0.f;

    for (int ss = 0; ss < TT; ss += 32) {
        const __bf16* ap = &pr[lhalf][ss + lsel * 8];
        v16bf af = make_frag(*(const uint4*)ap, *(const uint4*)(ap + 16));
#pragma unroll
        for (int nt = 0; nt < 8; ++nt) {
            const __bf16* bp =
                Vbase + (size_t)(dbase + nt * 16 + lhalf) * TT + ss + lsel * 8;
            v16bf bfr = make_frag(*(const uint4*)bp, *(const uint4*)(bp + 16));
            oacc[nt] = wmma_bf16f32(af, bfr, oacc[nt]);
        }
    }
    float* obase = out + ((size_t)b * TT + t0) * DD;
#pragma unroll
    for (int nt = 0; nt < 8; ++nt) {
        const int d = dbase + nt * 16 + lhalf;
        const float g = gate[(size_t)b * DD + d];
#pragma unroll
        for (int v = 0; v < 8; ++v)
            obase[(size_t)(v + lsel * 8) * DD + d] = g * oacc[nt][v];
    }
}

// ---------------------------------------------------------------------------
extern "C" void kernel_launch(void* const* d_in, const int* in_sizes, int n_in,
                              void* d_out, int out_size, void* d_ws,
                              size_t ws_size, hipStream_t stream) {
    const float* x        = (const float*)d_in[0];
    const int*   subj_pos = (const int*)d_in[1];
    /* obj_pos (d_in[2]) unused: reference pools object with subj_mask too */
    const int*   masks    = (const int*)d_in[3];
    const float* Wq = (const float*)d_in[4];
    const float* bq = (const float*)d_in[5];
    const float* Wc = (const float*)d_in[6];
    const float* bc = (const float*)d_in[7];
    const float* Wk = (const float*)d_in[8];
    const float* bk = (const float*)d_in[9];
    const float* Wm = (const float*)d_in[10];
    const float* bm = (const float*)d_in[11];
    const float* WK = (const float*)d_in[12];
    const float* bK = (const float*)d_in[13];
    const float* WQ = (const float*)d_in[14];
    const float* bQ = (const float*)d_in[15];
    const float* WV = (const float*)d_in[16];
    const float* bV = (const float*)d_in[17];

    // output layout: [output | masks | gcn_inputs | att]  (flat f32)
    float* out0 = (float*)d_out;                        // [B,T,D]
    float* out1 = out0 + (size_t)BB * TT * DD;          // [B,T]
    float* out2 = out1 + (size_t)BB * TT;               // [B,T,D]
    float* out3 = out2 + (size_t)BB * TT * DD;          // [B,T]

    // workspace carve-up (~330 MB total)
    char*  ws  = (char*)d_ws;
    size_t off = 0;
    auto carve = [&](size_t bytes) -> void* {
        void* p = ws + off;
        off = (off + bytes + 255) & ~(size_t)255;
        return p;
    };
    __bf16* Xbf = (__bf16*)carve((size_t)MM * DD * 2);
    __bf16* WKt = (__bf16*)carve((size_t)DD * DD * 2);
    __bf16* WQt = (__bf16*)carve((size_t)DD * DD * 2);
    __bf16* WVt = (__bf16*)carve((size_t)DD * DD * 2);
    __bf16* Kbf = (__bf16*)carve((size_t)MM * DD * 2);
    __bf16* Qbf = (__bf16*)carve((size_t)MM * DD * 2);
    __bf16* Vbf = (__bf16*)carve((size_t)MM * DD * 2);
    __bf16* Vtb = (__bf16*)carve((size_t)MM * DD * 2);
    float* so   = (float*)carve((size_t)BB * 2 * DD * 4);
    float* qb   = (float*)carve((size_t)BB * DD * 4);
    float* twb  = (float*)carve((size_t)BB * DD * 4);
    float* klog = (float*)carve((size_t)BB * TT * 4);
    float* kw   = (float*)carve((size_t)BB * TT * 4);
    float* cb   = (float*)carve((size_t)BB * DD * 4);
    float* mb   = (float*)carve((size_t)BB * DD * 4);
    (void)ws_size; (void)n_in; (void)in_sizes; (void)out_size;

    // fused X pass: bf16 convert + pass-through outputs
    xcvt_copy_kernel<<<MM * DD / (8 * 256), 256, 0, stream>>>(x, masks, Xbf,
                                                              out1, out2);

    // weight prep (transpose + bf16)
    wt_kernel<<<DD * DD / 256, 256, 0, stream>>>(WK, WKt);
    wt_kernel<<<DD * DD / 256, 256, 0, stream>>>(WQ, WQt);
    wt_kernel<<<DD * DD / 256, 256, 0, stream>>>(WV, WVt);

    // small serial chain
    pool_kernel<<<BB, 256, 0, stream>>>(x, subj_pos, so);
    q_kernel<<<dim3(DD / 256, BB), 256, 0, stream>>>(so, Wq, bq, qb);
    tw_kernel<<<dim3(DD / 256, BB), 256, 0, stream>>>(qb, Wc, bc, Wk, twb);
    klogit_kernel<<<MM / 8, 256, 0, stream>>>(x, twb, bk, klog);
    softmaxT_kernel<<<BB, 256, 0, stream>>>(klog, kw);
    c_kernel<<<dim3(DD / 256, BB), 256, 0, stream>>>(x, kw, cb);
    m_kernel<<<dim3(DD / 256, BB), 256, 0, stream>>>(cb, so, Wm, bm, mb);

    // big WMMA projections (TDM-fed): K/Q/V = X @ W + b (bf16 out)
    const dim3 ggrid(DD / 128, MM / 128);
    gemm_proj_kernel<<<ggrid, 256, 0, stream>>>(Xbf, WKt, bK, Kbf);
    gemm_proj_kernel<<<ggrid, 256, 0, stream>>>(Xbf, WQt, bQ, Qbf);
    gemm_proj_kernel<<<ggrid, 256, 0, stream>>>(Xbf, WVt, bV, Vbf);
    vt_kernel<<<MM * (DD / 256), 256, 0, stream>>>(Vbf, Vtb);

    // fused attention (scores -> dual softmax -> att + gated output)
    attn_kernel<<<dim3(TT / 16, BB), 256, 0, stream>>>(Kbf, Qbf, Vtb, mb,
                                                       out0, out3);
}
